// LSTM_24455543783997
// MI455X (gfx1250) — matrix operand order repaired and checked by reading, not compiled, not run on hardware
//
#include <hip/hip_runtime.h>

typedef __attribute__((ext_vector_type(16))) __bf16 v16bf;
typedef __attribute__((ext_vector_type(8)))  __bf16 v8bf;
typedef __attribute__((ext_vector_type(8)))  float  v8f;

#define T_STEPS 128
#define BATCH   64
#define HID     1024
#define NL      4
#define G4H     (4 * HID)

__device__ __forceinline__ float sigf(float x) {
    return 1.0f / (1.0f + __expf(-x));
}
__device__ __forceinline__ float tanhfast(float x) {
    x = fminf(fmaxf(x, -15.0f), 15.0f);
    float e = __expf(2.0f * x);
    return (e - 1.0f) / (e + 1.0f);
}

// One wave per 16x16 (batch x hidden) tile. Computes z = Ax@Wt^T + Ah@Ut^T + b
// for all 4 gates via v_wmma_f32_16x16x32_bf16, then the LSTM gate math in f32.
__global__ __launch_bounds__(32)
void lstm_step(const __bf16* __restrict__ Ax,     // [B,H] layer input (bf16)
               const __bf16* __restrict__ Ah,     // [B,H] h_prev (bf16)
               const __bf16* __restrict__ Wt,     // [4H,H] bf16, pre-transposed
               const __bf16* __restrict__ Ut,     // [4H,H] bf16, pre-transposed
               const float*  __restrict__ bias,   // [4H]
               const float*  __restrict__ Cprev,  // [B,H] f32
               float*        __restrict__ Cnew,   // [B,H] f32
               __bf16*       __restrict__ Hnew,   // [B,H] bf16 (next t's A_h)
               __bf16*       __restrict__ CurOut) // [B,H] bf16 c (next layer in)
{
    const int lane = threadIdx.x;      // 0..31
    const int half = lane >> 4;        // 0 or 1
    const int l16  = lane & 15;
    const int n0 = blockIdx.x * 16;    // hidden tile base
    const int m0 = blockIdx.y * 16;    // batch tile base

    v8f acc[4] = {};                   // gates i,f,g,o

    const __bf16* axrow = Ax + (size_t)(m0 + l16) * HID;
    const __bf16* ahrow = Ah + (size_t)(m0 + l16) * HID;

    union AFrag { v16bf v; v8bf h[2]; };

    for (int k0 = 0; k0 < HID; k0 += 32) {
        // A k order used here (contiguous, not the ISA table order): lane<16 holds K k0..k0+15;
        // lane>=16 holds k0+16..k0+31 (matches B).
        const int kk = k0 + half * 16;
        AFrag ax, ah;
        ax.h[0] = *(const v8bf*)(axrow + kk);
        ax.h[1] = *(const v8bf*)(axrow + kk + 8);
        ah.h[0] = *(const v8bf*)(ahrow + kk);
        ah.h[1] = *(const v8bf*)(ahrow + kk + 8);

        // B k order used here (contiguous, same as A, not the ISA table order): lanes 0-15 hold K k0..k0+15 at
        // column N=lane; lanes 16-31 hold K k0+16..k0+31. Weights are
        // pre-transposed to [4H][H] so each lane loads 16 contiguous bf16.
        const int kb = k0 + half * 16;
#pragma unroll
        for (int g = 0; g < 4; ++g) {
            const size_t col = (size_t)g * HID + n0 + l16;
            v16bf bw = *(const v16bf*)(Wt + col * HID + kb);
            v16bf bu = *(const v16bf*)(Ut + col * HID + kb);
            acc[g] = __builtin_amdgcn_wmma_f32_16x16x32_bf16(
                false, ax.v, false, bw, (short)0, acc[g], false, false);
            acc[g] = __builtin_amdgcn_wmma_f32_16x16x32_bf16(
                false, ah.v, false, bu, (short)0, acc[g], false, false);
        }
    }

    // C/D layout: VGPR r, lane -> M = m0 + r + 8*half, N = n0 + (lane&15)
    const int n = n0 + l16;
    const float bi = bias[0 * HID + n];
    const float bf = bias[1 * HID + n];
    const float bg = bias[2 * HID + n];
    const float bo = bias[3 * HID + n];

#pragma unroll
    for (int r = 0; r < 8; ++r) {
        const int m = m0 + r + half * 8;
        const float zi = acc[0][r] + bi;
        const float zf = acc[1][r] + bf;
        const float zg = acc[2][r] + bg;
        const float zo = acc[3][r] + bo;
        const float cp = Cprev[(size_t)m * HID + n];
        const float c  = sigf(zf) * cp + sigf(zi) * tanhfast(zg);
        const float h  = sigf(zo) * tanhfast(c);
        Cnew[(size_t)m * HID + n]   = c;
        Hnew[(size_t)m * HID + n]   = (__bf16)h;
        CurOut[(size_t)m * HID + n] = (__bf16)c;
    }
}

__global__ void cvt_f32_bf16(const float* __restrict__ src,
                             __bf16* __restrict__ dst, int n) {
    int i = blockIdx.x * blockDim.x + threadIdx.x;
    if (i < n) dst[i] = (__bf16)src[i];
}

// W,U: [L][H][4H] f32 -> Wt,Ut: [L][4H][H] bf16 (transpose + downconvert)
__global__ void transpose_wu(const float* __restrict__ W,
                             const float* __restrict__ U,
                             __bf16* __restrict__ Wt,
                             __bf16* __restrict__ Ut) {
    const int per_layer = HID * G4H;
    int i = blockIdx.x * blockDim.x + threadIdx.x;
    if (i >= NL * per_layer) return;
    int l   = i / per_layer;
    int rem = i - l * per_layer;
    int nn  = rem / HID;        // output row 0..4095
    int k   = rem - nn * HID;   // 0..1023
    size_t src = (size_t)l * per_layer + (size_t)k * G4H + nn;
    Wt[i] = (__bf16)W[src];
    Ut[i] = (__bf16)U[src];
}

__global__ void zero_state(__bf16* __restrict__ h0, float* __restrict__ c0, int n) {
    int i = blockIdx.x * blockDim.x + threadIdx.x;
    if (i < n) { h0[i] = (__bf16)0.0f; c0[i] = 0.0f; }
}

extern "C" void kernel_launch(void* const* d_in, const int* in_sizes, int n_in,
                              void* d_out, int out_size, void* d_ws, size_t ws_size,
                              hipStream_t stream) {
    (void)in_sizes; (void)n_in; (void)out_size;
    const float* x = (const float*)d_in[0];  // [T,B,H]
    const float* W = (const float*)d_in[1];  // [L,H,4H]
    const float* U = (const float*)d_in[2];  // [L,H,4H]
    const float* b = (const float*)d_in[3];  // [L,4H]
    float* out = (float*)d_out;              // [B,H]

    const size_t BH   = (size_t)BATCH * HID;
    const size_t LBH  = (size_t)NL * BH;
    const size_t WSZ  = (size_t)NL * G4H * HID;          // elems per weight tensor

    char* ws = (char*)d_ws;
    size_t off = 0;
    __bf16* Wt  = (__bf16*)(ws + off); off += WSZ * sizeof(__bf16);
    __bf16* Ut  = (__bf16*)(ws + off); off += WSZ * sizeof(__bf16);
    __bf16* xbf = (__bf16*)(ws + off); off += (size_t)T_STEPS * BH * sizeof(__bf16);
    __bf16* hbf = (__bf16*)(ws + off); off += 2 * LBH * sizeof(__bf16); // double buf
    float*  c32 = (float*) (ws + off); off += 2 * LBH * sizeof(float);  // double buf
    __bf16* cbf = (__bf16*)(ws + off); off += LBH * sizeof(__bf16);
    if (off > ws_size) return;  // workspace too small: bail safely

    {   // one-time (per call) prep: convert inputs, transpose+convert weights
        int nx = T_STEPS * BATCH * HID;
        cvt_f32_bf16<<<(nx + 255) / 256, 256, 0, stream>>>(x, xbf, nx);
        int nw = NL * HID * G4H;
        transpose_wu<<<(nw + 255) / 256, 256, 0, stream>>>(W, U, Wt, Ut);
        int ns = (int)LBH;
        zero_state<<<(ns + 255) / 256, 256, 0, stream>>>(hbf, c32, ns);
    }

    dim3 grid(HID / 16, BATCH / 16);  // 64 x 4 = 256 waves per step
    for (int t = 0; t < T_STEPS; ++t) {
        const int r = t & 1;          // read buffer (state from t-1)
        const int w = 1 - r;          // write buffer
        for (int j = 0; j < NL; ++j) {
            const __bf16* Ax = (j == 0) ? (xbf + (size_t)t * BH)
                                        : (cbf + (size_t)(j - 1) * BH);
            lstm_step<<<grid, 32, 0, stream>>>(
                Ax,
                hbf + (size_t)r * LBH + (size_t)j * BH,
                Wt  + (size_t)j * G4H * HID,
                Ut  + (size_t)j * G4H * HID,
                b   + (size_t)j * G4H,
                c32 + (size_t)r * LBH + (size_t)j * BH,
                c32 + (size_t)w * LBH + (size_t)j * BH,
                hbf + (size_t)w * LBH + (size_t)j * BH,
                cbf + (size_t)j * BH);
        }
    }

    // reference returns bb[L-1]: final cell state of last layer (f32)
    const int wf = 1 - ((T_STEPS - 1) & 1);
    hipMemcpyAsync(out, c32 + (size_t)wf * LBH + (size_t)(NL - 1) * BH,
                   BH * sizeof(float), hipMemcpyDeviceToDevice, stream);
}
